// SnapshotTemporalGNN_90220083020153
// MI455X (gfx1250) — matrix operand order repaired and checked by reading, not compiled, not run on hardware
//
#include <hip/hip_runtime.h>
#include <hip/hip_bf16.h>

#define T_STEPS 8
#define NNODES  50000
#define NEDGES  800000
#define IN_DIM  128
#define HID     128

typedef __attribute__((ext_vector_type(16))) __bf16 v16bf;
typedef __attribute__((ext_vector_type(8)))  float  v8f;

__device__ __forceinline__ float fixnum(float x) {
  if (x != x) return 0.f;
  if (x == __builtin_inff()) return 5.f;
  if (x == -__builtin_inff()) return -5.f;
  return x;
}

// |x| via explicit v_and_b32: opaque to the SLP vectorizer, which otherwise
// forms a packed-fabs bitcast pattern the gfx1250 ISel cannot select.
__device__ __forceinline__ float fabs_asm(float x) {
  float r;
  asm volatile("v_and_b32 %0, 0x7fffffff, %1" : "=v"(r) : "v"(x));
  return r;
}

// ---------------------------------------------------------------------------
// Weight pre-pack: fp32 -> bf16 in exact WMMA B-fragment order.
// Packed layout: [nt][kcid][lane][16 bf16], nt = column tile (16 cols),
// kcid = 32-wide K chunk. Fragment j-pair (2j,2j+1) holds B[kk],B[kk+1] with
// kk = kcid*32 + (j>>2)*16 + (lane>>4)*8 + 2*(j&3), col = nt*16 + (lane&15).
// trans=0: B is [K x ncols] row-major (ldb=ncols). trans=1: B is [ncols x K].
// ---------------------------------------------------------------------------
__global__ void pack_b_kernel(const float* __restrict__ B, int ldb, int K,
                              int ncols, int trans, __bf16* __restrict__ out) {
  int idx = blockIdx.x * blockDim.x + threadIdx.x;  // one per j-pair
  int kchunks = K >> 5;
  int total = (ncols >> 4) * kchunks * 32 * 8;
  if (idx >= total) return;
  int j    = idx & 7;
  int lane = (idx >> 3) & 31;
  int kcid = (idx >> 8) % kchunks;
  int nt   = (idx >> 8) / kchunks;
  int kk   = (kcid << 5) + ((j >> 2) << 4) + ((lane >> 4) << 3) + ((j & 3) << 1);
  int col  = (nt << 4) + (lane & 15);
  float y0, y1;
  if (trans) {
    y0 = B[(size_t)col * ldb + kk];
    y1 = B[(size_t)col * ldb + kk + 1];
  } else {
    y0 = B[(size_t)kk * ldb + col];
    y1 = B[(size_t)(kk + 1) * ldb + col];
  }
  out[2 * idx]     = (__bf16)y0;
  out[2 * idx + 1] = (__bf16)y1;
}

// ---------------------------------------------------------------------------
// A-fragment build: 16-bit A 16x32 layout = two contiguous 8-float runs per
// lane (offsets [0..7] and [16..23] from Ar = row + kc + half*8).
// ---------------------------------------------------------------------------
template <int ATR>
__device__ __forceinline__ v16bf make_a_frag(const float* __restrict__ Ar) {
  float vals[16];
  *(float4*)(vals + 0)  = *(const float4*)(Ar + 0);
  *(float4*)(vals + 4)  = *(const float4*)(Ar + 4);
  *(float4*)(vals + 8)  = *(const float4*)(Ar + 16);
  *(float4*)(vals + 12) = *(const float4*)(Ar + 20);
  v16bf a;
#pragma unroll
  for (int i = 0; i < 16; ++i) {
    float x = vals[i];
    if (ATR == 1) x = fmaxf(x, 0.f);
    if (ATR == 2) x = fixnum(x);
    a[i] = (__bf16)x;
  }
  return a;
}

// ---------------------------------------------------------------------------
// WMMA GEMM: C[M x ncols] = A[M x K] @ B + bias, B pre-packed bf16.
// Wave = one 16-row tile x 128 columns. blockIdx.y = 128-column group.
// Packed weights for this column group (<=32KB for K=128) are staged in LDS.
// ATR: 0 identity, 1 relu, 2 nan_to_num(0,+5,-5)
// ---------------------------------------------------------------------------
template <int ATR>
__global__ __launch_bounds__(256) void wmma_gemm_kernel(
    const float* __restrict__ A, int lda,
    const __bf16* __restrict__ Bp,
    const float* __restrict__ bias,
    float* __restrict__ C, int ldc,
    int Mtiles, int K) {
  __shared__ __align__(16) __bf16 smem[8 * 4 * 32 * 16];  // 32KB (K<=128)
  const int kchunks = K >> 5;

  // cooperative stage of this column group's packed weights into LDS
  {
    const size_t regionElems = (size_t)8 * kchunks * 32 * 16;
    const uint4* s4 = (const uint4*)(Bp + (size_t)blockIdx.y * regionElems);
    uint4* d4 = (uint4*)smem;
    const int n16 = (int)(regionElems >> 3);  // 8 bf16 per uint4
    for (int i = threadIdx.x; i < n16; i += 256) d4[i] = s4[i];
  }
  __syncthreads();

  const int wave = threadIdx.x >> 5;
  const int lane = threadIdx.x & 31;
  const int tile = blockIdx.x * 8 + wave;
  if (tile >= Mtiles) return;  // wave-uniform, after the block barrier
  const int mbase = tile << 4;
  const int half  = lane >> 4;
  const int l15   = lane & 15;
  const int col0  = blockIdx.y * 128;

  v8f acc[8] = {};

  const float* Arow = A + (size_t)(mbase + l15) * lda + (half << 3);
  for (int kcid = 0; kcid < kchunks; ++kcid) {
    v16bf a = make_a_frag<ATR>(Arow + (kcid << 5));
#pragma unroll
    for (int nt = 0; nt < 8; ++nt) {
      const v16bf b =
          *(const v16bf*)(smem + ((((nt * kchunks + kcid) << 5) + lane) << 4));
      acc[nt] = __builtin_amdgcn_wmma_f32_16x16x32_bf16(
          false, a, false, b, (short)0, acc[nt], false, false);
    }
  }

#pragma unroll
  for (int nt = 0; nt < 8; ++nt) {
    const int col = col0 + (nt << 4) + l15;
    const float bv = bias ? bias[col] : 0.f;
#pragma unroll
    for (int r = 0; r < 8; ++r) {
      C[(size_t)(mbase + r + (half << 3)) * ldc + col] = acc[nt][r] + bv;
    }
  }
}

// ---------------------------------------------------------------------------
// GCN degree + scatter
// ---------------------------------------------------------------------------
__global__ void deg_init_kernel(float* deg, int n) {
  int i = blockIdx.x * blockDim.x + threadIdx.x;
  if (i < n) deg[i] = 1.0f;  // self loop
}

__global__ void deg_acc_kernel(const int* __restrict__ dst, float* deg, int e) {
  int i = blockIdx.x * blockDim.x + threadIdx.x;
  if (i < e) atomicAdd(&deg[dst[i]], 1.0f);
}

__global__ void dinv_kernel(float* deg, int n) {
  int i = blockIdx.x * blockDim.x + threadIdx.x;
  if (i < n) deg[i] = __frsqrt_rn(deg[i]);  // deg >= 1 always
}

__global__ void scatter_init_kernel(const float* __restrict__ h,
                                    const float* __restrict__ dinv,
                                    const float* __restrict__ bias,
                                    float* __restrict__ out, int n) {
  int idx = blockIdx.x * blockDim.x + threadIdx.x;
  if (idx >= n * HID) return;
  int i = idx >> 7, f = idx & 127;
  float di = dinv[i];
  out[idx] = bias[f] + di * di * h[idx];
}

__global__ __launch_bounds__(256) void scatter_edge_kernel(
    const float* __restrict__ h, const int* __restrict__ src,
    const int* __restrict__ dst, const float* __restrict__ dinv,
    float* __restrict__ out, int e) {
  int w = (blockIdx.x * blockDim.x + threadIdx.x) >> 5;
  int lane = threadIdx.x & 31;
  if (w >= e) return;
  int s = src[w], d = dst[w];
  float wt = dinv[s] * dinv[d];
  const float4 hv = *(const float4*)(h + (size_t)s * HID + lane * 4);
  float* op = out + (size_t)d * HID + lane * 4;
  atomicAdd(op + 0, hv.x * wt);
  atomicAdd(op + 1, hv.y * wt);
  atomicAdd(op + 2, hv.z * wt);
  atomicAdd(op + 3, hv.w * wt);
}

// ---------------------------------------------------------------------------
// GRU gate fusion
// ---------------------------------------------------------------------------
__device__ __forceinline__ float sigmoidf(float x) { return 1.f / (1.f + __expf(-x)); }

__global__ void gru_gate_kernel(const float* __restrict__ gi,
                                const float* __restrict__ gh,
                                const float* __restrict__ hprev,
                                float* __restrict__ hout, int n) {
  int idx = blockIdx.x * blockDim.x + threadIdx.x;
  if (idx >= n * HID) return;
  int i = idx >> 7, f = idx & 127;
  const float* gir = gi + (size_t)i * 384;
  const float* ghr = gh + (size_t)i * 384;
  float r  = sigmoidf(gir[f] + ghr[f]);
  float z  = sigmoidf(gir[128 + f] + ghr[128 + f]);
  float nn = tanhf(gir[256 + f] + r * ghr[256 + f]);
  hout[idx] = (1.f - z) * nn + z * hprev[idx];
}

// ---------------------------------------------------------------------------
// Attention
// ---------------------------------------------------------------------------
__global__ __launch_bounds__(256) void att_score_kernel(
    const float* __restrict__ H, const float* __restrict__ attW,
    const float* __restrict__ attB, float* __restrict__ att, int rows) {
  int w = (blockIdx.x * blockDim.x + threadIdx.x) >> 5;
  int lane = threadIdx.x & 31;
  if (w >= rows) return;
  const float4 hv = *(const float4*)(H + (size_t)w * HID + lane * 4);
  const float4 wv = *(const float4*)(attW + lane * 4);
  float p = hv.x * wv.x + hv.y * wv.y + hv.z * wv.z + hv.w * wv.w;
#pragma unroll
  for (int off = 16; off; off >>= 1) p += __shfl_xor(p, off, 32);
  if (lane == 0) {
    float a = p + attB[0];
    att[w] = fminf(fmaxf(a, -10.f), 10.f);
  }
}

__global__ __launch_bounds__(256) void att_sum_kernel(
    const float* __restrict__ H, const float* __restrict__ att,
    float* __restrict__ z, int n) {
  int w = (blockIdx.x * blockDim.x + threadIdx.x) >> 5;
  int lane = threadIdx.x & 31;
  if (w >= n) return;
  float a[T_STEPS];
  float mx = -3.0e38f;
#pragma unroll
  for (int t = 0; t < T_STEPS; ++t) { a[t] = att[(size_t)t * n + w]; mx = fmaxf(mx, a[t]); }
  float s = 0.f;
#pragma unroll
  for (int t = 0; t < T_STEPS; ++t) { a[t] = __expf(a[t] - mx); s += a[t]; }
  float inv = 1.f / s;
  float4 acc = make_float4(0.f, 0.f, 0.f, 0.f);
#pragma unroll
  for (int t = 0; t < T_STEPS; ++t) {
    const float4 hv = *(const float4*)(H + ((size_t)t * n + w) * HID + lane * 4);
    float wt = a[t] * inv;
    acc.x += hv.x * wt; acc.y += hv.y * wt; acc.z += hv.z * wt; acc.w += hv.w * wt;
  }
  acc.x = fixnum(acc.x); acc.y = fixnum(acc.y); acc.z = fixnum(acc.z); acc.w = fixnum(acc.w);
  *(float4*)(z + (size_t)w * HID + lane * 4) = acc;
}

// ---------------------------------------------------------------------------
// Decoder: WMMA GEMM over on-the-fly features [s, d, |s-d|, s*d] (K=512),
// fused leaky-ReLU + final 128->1 dot in the epilogue -> logits directly.
// Each 32-wide K chunk sits in one 128-feature block, so the selector is
// wave-uniform and the z-row loads are float4.
// ---------------------------------------------------------------------------
__global__ __launch_bounds__(256) void decoder_kernel(
    const float* __restrict__ z, const int* __restrict__ esrc,
    const int* __restrict__ edst, const __bf16* __restrict__ W1p,
    const float* __restrict__ b1, const float* __restrict__ w2,
    const float* __restrict__ b2, float* __restrict__ logits, int Etiles) {
  const int wave = threadIdx.x >> 5;
  const int lane = threadIdx.x & 31;
  const int tile = blockIdx.x * 8 + wave;
  if (tile >= Etiles) return;
  const int half = lane >> 4;
  const int l15  = lane & 15;
  const int e    = (tile << 4) + l15;
  const int s = esrc[e], d = edst[e];
  const float* zs = z + (size_t)s * HID + (half << 3);
  const float* zd = z + (size_t)d * HID + (half << 3);

  v8f acc[8] = {};

  for (int kc = 0; kc < 512; kc += 32) {
    const int sel = kc >> 7;
    const int fb  = kc & 127;
    float vs[16], vd[16];
    if (sel != 1) {
      *(float4*)(vs + 0)  = *(const float4*)(zs + fb + 0);
      *(float4*)(vs + 4)  = *(const float4*)(zs + fb + 4);
      *(float4*)(vs + 8)  = *(const float4*)(zs + fb + 16);
      *(float4*)(vs + 12) = *(const float4*)(zs + fb + 20);
    }
    if (sel != 0) {
      *(float4*)(vd + 0)  = *(const float4*)(zd + fb + 0);
      *(float4*)(vd + 4)  = *(const float4*)(zd + fb + 4);
      *(float4*)(vd + 8)  = *(const float4*)(zd + fb + 16);
      *(float4*)(vd + 12) = *(const float4*)(zd + fb + 20);
    }
    v16bf a;
#pragma unroll
    for (int i = 0; i < 16; ++i) {
      float x;
      if (sel == 0)      x = vs[i];
      else if (sel == 1) x = vd[i];
      else if (sel == 2) x = fabs_asm(vs[i] - vd[i]);
      else               x = vs[i] * vd[i];
      a[i] = (__bf16)x;
    }
    const int kcid = kc >> 5;
#pragma unroll
    for (int nt = 0; nt < 8; ++nt) {
      const v16bf b =
          *(const v16bf*)(W1p + ((((size_t)(nt * 16 + kcid) << 5) + lane) << 4));
      acc[nt] = __builtin_amdgcn_wmma_f32_16x16x32_bf16(
          false, a, false, b, (short)0, acc[nt], false, false);
    }
  }

  // fused: leaky_relu(hm + b1) . w2, reduced across the 16 lanes of each half
  float partial[8];
#pragma unroll
  for (int r = 0; r < 8; ++r) partial[r] = 0.f;
#pragma unroll
  for (int nt = 0; nt < 8; ++nt) {
    const int col = (nt << 4) + l15;
    const float bb = b1[col], ww = w2[col];
#pragma unroll
    for (int r = 0; r < 8; ++r) {
      float v = acc[nt][r] + bb;
      v = v > 0.f ? v : 0.1f * v;
      partial[r] += v * ww;
    }
  }
#pragma unroll
  for (int r = 0; r < 8; ++r) {
    float p = partial[r];
    p += __shfl_xor(p, 1, 16);
    p += __shfl_xor(p, 2, 16);
    p += __shfl_xor(p, 4, 16);
    p += __shfl_xor(p, 8, 16);
    partial[r] = p;
  }
  if (l15 == 0) {
    const float bb2 = b2[0];
    const int rowbase = (tile << 4) + (half << 3);
#pragma unroll
    for (int r = 0; r < 8; ++r) logits[rowbase + r] = partial[r] + bb2;
  }
}

// ---------------------------------------------------------------------------
// Orchestration
// ---------------------------------------------------------------------------
extern "C" void kernel_launch(void* const* d_in, const int* in_sizes, int n_in,
                              void* d_out, int out_size, void* d_ws, size_t ws_size,
                              hipStream_t stream) {
  const float* X      = (const float*)d_in[0];   // [T,N,128]
  const int*   eiSeq  = (const int*)  d_in[1];   // [T,2,E]
  const int*   eiDec  = (const int*)  d_in[2];   // [2,E]
  const float* projW  = (const float*)d_in[3];
  const float* projB  = (const float*)d_in[4];
  const float* gW1    = (const float*)d_in[5];
  const float* gB1    = (const float*)d_in[6];
  const float* gW2    = (const float*)d_in[7];
  const float* gB2    = (const float*)d_in[8];
  const float* Wih    = (const float*)d_in[9];   // [384,128]
  const float* Whh    = (const float*)d_in[10];  // [384,128]
  const float* bih    = (const float*)d_in[11];
  const float* bhh    = (const float*)d_in[12];
  const float* attW   = (const float*)d_in[13];  // [128,1]
  const float* attB   = (const float*)d_in[14];
  const float* mW1    = (const float*)d_in[15];  // [512,128]
  const float* mB1    = (const float*)d_in[16];
  const float* mW2    = (const float*)d_in[17];  // [128,1]
  const float* mB2    = (const float*)d_in[18];
  float* logits = (float*)d_out;

  const int N = NNODES, E = NEDGES;

  // workspace carving (fp32 part)
  float* ws = (float*)d_ws;
  size_t off = 0;
  float* tmp1 = ws + off; off += (size_t)N * HID;
  float* tmp2 = ws + off; off += (size_t)N * HID;
  float* deg  = ws + off; off += (size_t)N;          // becomes dinv in place
  float* gi   = ws + off; off += (size_t)N * 384;
  float* gh   = ws + off; off += (size_t)N * 384;
  float* Htmp = ws + off; off += (size_t)T_STEPS * N * HID;
  float* attS = ws + off; off += (size_t)T_STEPS * N;
  float* zbuf = ws + off; off += (size_t)N * HID;
  float* h0   = ws + off; off += (size_t)N * HID;

  // packed bf16 weights (all 16B-aligned: every fp32 offset above is /16)
  __bf16* wb = (__bf16*)(ws + off);
  size_t boff = 0;
  __bf16* projWp = wb + boff; boff += (size_t)8 * 4 * 32 * 16;   // 128c,K128
  __bf16* gW1p   = wb + boff; boff += (size_t)8 * 4 * 32 * 16;
  __bf16* gW2p   = wb + boff; boff += (size_t)8 * 4 * 32 * 16;
  __bf16* Wihp   = wb + boff; boff += (size_t)24 * 4 * 32 * 16;  // 384c,K128
  __bf16* Whhp   = wb + boff; boff += (size_t)24 * 4 * 32 * 16;
  __bf16* mW1p   = wb + boff; boff += (size_t)8 * 16 * 32 * 16;  // 128c,K512

  (void)hipMemsetAsync(h0, 0, (size_t)N * HID * sizeof(float), stream);

  const dim3 blk(256);

  // pack all weights (tiny)
  pack_b_kernel<<<(8 * 4 * 32 * 8 + 255) / 256, blk, 0, stream>>>(
      projW, HID, 128, 128, 0, projWp);
  pack_b_kernel<<<(8 * 4 * 32 * 8 + 255) / 256, blk, 0, stream>>>(
      gW1, HID, 128, 128, 0, gW1p);
  pack_b_kernel<<<(8 * 4 * 32 * 8 + 255) / 256, blk, 0, stream>>>(
      gW2, HID, 128, 128, 0, gW2p);
  pack_b_kernel<<<(24 * 4 * 32 * 8 + 255) / 256, blk, 0, stream>>>(
      Wih, HID, 128, 384, 1, Wihp);
  pack_b_kernel<<<(24 * 4 * 32 * 8 + 255) / 256, blk, 0, stream>>>(
      Whh, HID, 128, 384, 1, Whhp);
  pack_b_kernel<<<(8 * 16 * 32 * 8 + 255) / 256, blk, 0, stream>>>(
      mW1, 128, 512, 128, 0, mW1p);

  const int Mtiles = N / 16;                       // 3125
  const int gemmBlocks = (Mtiles + 7) / 8;
  const int ewBlocks   = (int)(((size_t)E * 32 + 255) / 256);
  const int nfBlocks   = (int)(((size_t)N * HID + 255) / 256);

  for (int t = 0; t < T_STEPS; ++t) {
    const float* Xt  = X + (size_t)t * N * IN_DIM;
    const int*   src = eiSeq + (size_t)t * 2 * E;
    const int*   dst = src + E;

    // h = X_t @ proj_W + proj_b
    wmma_gemm_kernel<0><<<dim3(gemmBlocks, 1), blk, 0, stream>>>(
        Xt, IN_DIM, projWp, projB, tmp1, HID, Mtiles, IN_DIM);

    // degrees (shared by both convs of this snapshot)
    deg_init_kernel<<<(N + 255) / 256, blk, 0, stream>>>(deg, N);
    deg_acc_kernel<<<(E + 255) / 256, blk, 0, stream>>>(dst, deg, E);
    dinv_kernel<<<(N + 255) / 256, blk, 0, stream>>>(deg, N);

    // GCN layer 1: hw = h @ W1 ; scatter with norm + b1
    wmma_gemm_kernel<0><<<dim3(gemmBlocks, 1), blk, 0, stream>>>(
        tmp1, HID, gW1p, nullptr, tmp2, HID, Mtiles, HID);
    scatter_init_kernel<<<nfBlocks, blk, 0, stream>>>(tmp2, deg, gB1, tmp1, N);
    scatter_edge_kernel<<<ewBlocks, blk, 0, stream>>>(tmp2, src, dst, deg, tmp1, E);

    // GCN layer 2: hw = relu(h) @ W2 (relu fused into A-load); scatter + b2
    wmma_gemm_kernel<1><<<dim3(gemmBlocks, 1), blk, 0, stream>>>(
        tmp1, HID, gW2p, nullptr, tmp2, HID, Mtiles, HID);
    scatter_init_kernel<<<nfBlocks, blk, 0, stream>>>(tmp2, deg, gB2, tmp1, N);
    scatter_edge_kernel<<<ewBlocks, blk, 0, stream>>>(tmp2, src, dst, deg, tmp1, E);

    // GRU step: gi = nan_to_num(h_emb) @ Wih^T + bih ; gh = h_prev @ Whh^T + bhh
    wmma_gemm_kernel<2><<<dim3(gemmBlocks, 3), blk, 0, stream>>>(
        tmp1, HID, Wihp, bih, gi, 384, Mtiles, HID);
    const float* hprev = (t == 0) ? h0 : (Htmp + (size_t)(t - 1) * N * HID);
    wmma_gemm_kernel<0><<<dim3(gemmBlocks, 3), blk, 0, stream>>>(
        hprev, HID, Whhp, bhh, gh, 384, Mtiles, HID);
    gru_gate_kernel<<<nfBlocks, blk, 0, stream>>>(
        gi, gh, hprev, Htmp + (size_t)t * N * HID, N);
  }

  // attention over time
  const int rows = T_STEPS * N;
  att_score_kernel<<<(int)(((size_t)rows * 32 + 255) / 256), blk, 0, stream>>>(
      Htmp, attW, attB, attS, rows);
  att_sum_kernel<<<(int)(((size_t)N * 32 + 255) / 256), blk, 0, stream>>>(
      Htmp, attS, zbuf, N);

  // fused decoder (feature build + GEMM + leaky relu + final dot)
  const int Etiles = E / 16;                        // 50000
  decoder_kernel<<<(Etiles + 7) / 8, blk, 0, stream>>>(
      zbuf, eiDec, eiDec + E, mW1p, mB1, mW2, mB2, logits, Etiles);
}
